// RMSNorm_72696616452309
// MI455X (gfx1250) — matrix operand order repaired and checked by reading, not compiled, not run on hardware
//
#include <hip/hip_runtime.h>

typedef __attribute__((ext_vector_type(16))) __bf16 v16bf;
typedef __attribute__((ext_vector_type(8)))  float  v8f;
typedef __attribute__((ext_vector_type(4)))  int    v4i_;

__device__ __forceinline__ unsigned short f32_to_bf16(float f) {
    unsigned int u = __float_as_uint(f);
    unsigned int r = u + 0x7FFFu + ((u >> 16) & 1u);   // round-to-nearest-even
    return (unsigned short)(r >> 16);
}

union FragB16 { v16bf v; uint4 q[2]; unsigned int u[8]; };

// ---------------------------------------------------------------------------
// CDNA5 async global->LDS copy (ASYNCcnt-tracked), with sync fallback
// ---------------------------------------------------------------------------
#if defined(__has_builtin)
#if __has_builtin(__builtin_amdgcn_global_load_async_to_lds_b128)
#define HAVE_ASYNC_LDS 1
#endif
#endif
#ifndef HAVE_ASYNC_LDS
#define HAVE_ASYNC_LDS 0
#endif

__device__ __forceinline__ void async_copy16(const unsigned short* g, unsigned short* l) {
#if HAVE_ASYNC_LDS
    __builtin_amdgcn_global_load_async_to_lds_b128((v4i_*)g, (v4i_*)l, 0, 0);
#else
    *(uint4*)l = *(const uint4*)g;
#endif
}

__device__ __forceinline__ void async_wait_all() {
#if HAVE_ASYNC_LDS
#if __has_builtin(__builtin_amdgcn_s_wait_asynccnt)
    __builtin_amdgcn_s_wait_asynccnt(0);
#else
    asm volatile("s_wait_asynccnt 0x0" ::: "memory");
#endif
#endif
}

// ---------------------------------------------------------------------------
// Elementwise f32 -> bf16 (vectorized 4/thread)
// ---------------------------------------------------------------------------
__global__ void cvt_bf16(const float* __restrict__ in, unsigned short* __restrict__ out, int n4) {
    int i = blockIdx.x * blockDim.x + threadIdx.x;
    if (i >= n4) return;
    float4 v = ((const float4*)in)[i];
    uint2 p;
    p.x = (unsigned)f32_to_bf16(v.x) | ((unsigned)f32_to_bf16(v.y) << 16);
    p.y = (unsigned)f32_to_bf16(v.z) | ((unsigned)f32_to_bf16(v.w) << 16);
    ((uint2*)out)[i] = p;
}

// ---------------------------------------------------------------------------
// Tiled GEMM: C[M,N] (f32) = A[M,K] * B[K,N], both bf16, via WMMA.
// 128x64x32 tiles, 8 waves, async double-buffered LDS staging.
// ---------------------------------------------------------------------------
constexpr int BM = 128, BN = 64, BK = 32;
constexpr int LDA_S = BK + 8;    // 40 halves = 80B rows (16B multiple)
constexpr int LDB_S = BN + 16;   // 80 halves = 160B rows

__global__ __launch_bounds__(256)
void gemm_bf16(const unsigned short* __restrict__ A, const unsigned short* __restrict__ B,
               float* __restrict__ C, int M, int N, int K) {
    __shared__ __attribute__((aligned(16))) unsigned short sA[2][BM * LDA_S];
    __shared__ __attribute__((aligned(16))) unsigned short sB[2][BK * LDB_S];

    const int tid  = threadIdx.x;
    const int lane = tid & 31;
    const int wave = tid >> 5;
    const int wm   = wave >> 1;          // 0..3 : 32-row strip
    const int wn   = wave & 1;           // 0..1 : 32-col strip
    const int m0   = blockIdx.y * BM;
    const int n0   = blockIdx.x * BN;
    const int lh   = lane >> 4;
    const int r16  = lane & 15;

    // per-thread staging coordinates (16B chunks)
    const int arow = tid >> 2, ac8 = (tid & 3) * 8;   // A: 128 rows x 4 chunks, 2 passes
    const int brow = tid >> 3, bc8 = (tid & 7) * 8;   // B: 32 rows x 8 chunks

    // prologue: stage k-tile 0 into buffer 0
    async_copy16(A + (size_t)(m0 + arow)      * K + ac8, &sA[0][arow * LDA_S + ac8]);
    async_copy16(A + (size_t)(m0 + arow + 64) * K + ac8, &sA[0][(arow + 64) * LDA_S + ac8]);
    async_copy16(B + (size_t)brow * N + n0 + bc8,        &sB[0][brow * LDB_S + bc8]);
    async_wait_all();
    __syncthreads();

    v8f acc[2][2] = {};
    int buf = 0;
    const int ko1 = lh ? 8 : 0, ko2 = lh ? 24 : 16;      // A-layout K split

    for (int kk = 0; kk < K; kk += BK) {
        // prefetch next k-tile into the other buffer (async, overlaps WMMA)
        if (kk + BK < K) {
            const int kn = kk + BK;
            async_copy16(A + (size_t)(m0 + arow)      * K + kn + ac8, &sA[buf ^ 1][arow * LDA_S + ac8]);
            async_copy16(A + (size_t)(m0 + arow + 64) * K + kn + ac8, &sA[buf ^ 1][(arow + 64) * LDA_S + ac8]);
            async_copy16(B + (size_t)(kn + brow) * N + n0 + bc8,      &sB[buf ^ 1][brow * LDB_S + bc8]);
        }

        // load all fragments, then do all WMMAs
        FragB16 af[2], bfr[2];
#pragma unroll
        for (int i = 0; i < 2; ++i) {
            int row = wm * 32 + i * 16 + r16;                // A: lane covers M
            af[i].q[0] = *(const uint4*)(&sA[buf][row * LDA_S + ko1]);
            af[i].q[1] = *(const uint4*)(&sA[buf][row * LDA_S + ko2]);
        }
#pragma unroll
        for (int j = 0; j < 2; ++j) {
            int col = wn * 32 + j * 16 + r16;                // B: lane covers N, elements cover K
#pragma unroll
            for (int w = 0; w < 8; ++w) {
                int k = (w < 4 ? 0 : 16) + 8 * lh + 2 * (w & 3);
                bfr[j].u[w] = (unsigned)sB[buf][k * LDB_S + col] | ((unsigned)sB[buf][(k + 1) * LDB_S + col] << 16);
            }
        }
#pragma unroll
        for (int i = 0; i < 2; ++i)
#pragma unroll
            for (int j = 0; j < 2; ++j)
                acc[i][j] = __builtin_amdgcn_wmma_f32_16x16x32_bf16(
                    false, af[i].v, false, bfr[j].v, (short)0, acc[i][j], false, false);

        async_wait_all();
        __syncthreads();
        buf ^= 1;
    }

    // epilogue: C layout = lane covers N, VGPR p covers M=p(+8*half)
#pragma unroll
    for (int i = 0; i < 2; ++i) {
        int rbase = m0 + wm * 32 + i * 16 + 8 * lh;
#pragma unroll
        for (int j = 0; j < 2; ++j) {
            int col = n0 + wn * 32 + j * 16 + r16;
#pragma unroll
            for (int p = 0; p < 8; ++p)
                C[(size_t)(rbase + p) * N + col] = acc[i][j][p];
        }
    }
}

// ---------------------------------------------------------------------------
// RoPE (+ head-major transpose) f32 [B,S,NH,64] -> bf16 [B,NH,S,64]
// ---------------------------------------------------------------------------
__global__ void rope_to_bf16(const float* __restrict__ x,
                             const float* __restrict__ cosb,
                             const float* __restrict__ sinb,
                             unsigned short* __restrict__ out,
                             int NH, int S, int do_rope, int total) {
    int i = blockIdx.x * blockDim.x + threadIdx.x;   // over B*S*NH*32 pairs
    if (i >= total) return;
    int d2 = i & 31;
    int h  = (i >> 5) % NH;
    int s  = (i / (32 * NH)) % S;
    int b  = i / (32 * NH * S);
    size_t in = ((size_t)(b * S + s) * NH + h) * 64 + 2 * d2;
    size_t oo = ((size_t)(b * NH + h) * S + s) * 64 + 2 * d2;
    float tr = x[in], ti = x[in + 1];
    if (do_rope) {
        float c = cosb[s * 32 + d2], sn = sinb[s * 32 + d2];
        float orr = tr * c - ti * sn;
        float oi  = tr * sn + ti * c;
        tr = orr; ti = oi;
    }
    out[oo]     = f32_to_bf16(tr);
    out[oo + 1] = f32_to_bf16(ti);
}

// ---------------------------------------------------------------------------
// Causal flash attention, bf16 WMMA, online softmax.
//   Q [B,H,S,64], K/V [B,KVH,S,64] (bf16) -> ctx [B,S,H*64] (bf16)
// ---------------------------------------------------------------------------
constexpr int AHD = 64, AH = 32, AKVH = 8, ANREP = 4;
constexpr int QB = 128, KBLK = 64;
constexpr int LKV = 80;                                  // 160B LDS rows

__global__ __launch_bounds__(256)
void flash_attn(const unsigned short* __restrict__ Q,
                const unsigned short* __restrict__ Kg,
                const unsigned short* __restrict__ Vg,
                unsigned short* __restrict__ Octx, int S) {
    __shared__ __attribute__((aligned(16))) unsigned short sKT[AHD * LKV];  // [hd][key]
    __shared__ __attribute__((aligned(16))) unsigned short sV [KBLK * LKV]; // [key][hd]
    __shared__ __attribute__((aligned(16))) unsigned short sP [8 * 16 * 32];// per-wave P

    const int q0  = blockIdx.x * QB;
    const int h   = blockIdx.y;
    const int b   = blockIdx.z;
    const int kvh = h / ANREP;
    const int tid = threadIdx.x, lane = tid & 31, wave = tid >> 5;
    const int lh = lane >> 4, r16 = lane & 15;

    const unsigned short* Qb = Q  + (size_t)(b * AH   + h  ) * S * AHD;
    const unsigned short* Kb = Kg + (size_t)(b * AKVH + kvh) * S * AHD;
    const unsigned short* Vb = Vg + (size_t)(b * AKVH + kvh) * S * AHD;

    // Q fragments for this wave's 16 rows, kept in registers all kernel
    FragB16 qf[2];
    {
        const unsigned short* rp = Qb + (size_t)(q0 + wave * 16 + r16) * AHD;
#pragma unroll
        for (int f = 0; f < 2; ++f) {
            int ko1 = f * 32 + (lh ? 8 : 0);
            int ko2 = f * 32 + (lh ? 24 : 16);
            qf[f].q[0] = *(const uint4*)(rp + ko1);
            qf[f].q[1] = *(const uint4*)(rp + ko2);
        }
    }

    float mrow[8], lrow[8];
#pragma unroll
    for (int p = 0; p < 8; ++p) { mrow[p] = -1e30f; lrow[p] = 0.f; }
    v8f o[4] = {};

    const int qwave_max = q0 + wave * 16 + 15;
    for (int kb = 0; kb < q0 + QB; kb += KBLK) {
        // ---- stage V [key][hd] (pure copy -> async to LDS) ----
        {
            int row = tid >> 2, c = (tid & 3) * 16;
            async_copy16(Vb + (size_t)(kb + row) * AHD + c,     &sV[row * LKV + c]);
            async_copy16(Vb + (size_t)(kb + row) * AHD + c + 8, &sV[row * LKV + c + 8]);
        }
        // ---- stage K^T [hd][key] (register transpose) ----
        {
            int key = tid & 63, hg = tid >> 6;               // hg: 16 hd-dims each
            const unsigned short* kp = Kb + (size_t)(kb + key) * AHD + hg * 16;
            uint4 a = *(const uint4*)(kp);
            uint4 c = *(const uint4*)(kp + 8);
            unsigned int w[8] = {a.x, a.y, a.z, a.w, c.x, c.y, c.z, c.w};
#pragma unroll
            for (int u = 0; u < 8; ++u) {
                sKT[(hg * 16 + 2 * u    ) * LKV + key] = (unsigned short)(w[u] & 0xFFFFu);
                sKT[(hg * 16 + 2 * u + 1) * LKV + key] = (unsigned short)(w[u] >> 16);
            }
        }
        async_wait_all();
        __syncthreads();

        if (kb <= qwave_max) {                                // causal wave-level skip
#pragma unroll
            for (int kc = 0; kc < KBLK; kc += 32) {
                // ---- S = Q K^T : preload all 4 B-fragments, then 4 WMMAs ----
                FragB16 kf[2][2];
#pragma unroll
                for (int j = 0; j < 2; ++j)
#pragma unroll
                    for (int f = 0; f < 2; ++f) {
                        int kcol = kc + j * 16 + r16;         // B: lane covers N(=key), elements cover K(=hd)
#pragma unroll
                        for (int w = 0; w < 8; ++w) {
                            int hd = f * 32 + (w < 4 ? 0 : 16) + 8 * lh + 2 * (w & 3);
                            kf[j][f].u[w] = (unsigned)sKT[hd * LKV + kcol] | ((unsigned)sKT[(hd + 1) * LKV + kcol] << 16);
                        }
                    }
                v8f sacc[2] = {};
#pragma unroll
                for (int j = 0; j < 2; ++j)
#pragma unroll
                    for (int f = 0; f < 2; ++f)
                        sacc[j] = __builtin_amdgcn_wmma_f32_16x16x32_bf16(
                            false, qf[f].v, false, kf[j][f].v, (short)0, sacc[j], false, false);

                // ---- online softmax over these 32 keys ----
                unsigned short* pp = &sP[wave * 16 * 32];
                const int qrow_base = q0 + wave * 16 + 8 * lh;
#pragma unroll
                for (int p = 0; p < 8; ++p) {
                    int qrow = qrow_base + p;
                    float s0 = fminf(fmaxf(sacc[0][p] * 0.125f, -20.f), 20.f);
                    float s1 = fminf(fmaxf(sacc[1][p] * 0.125f, -20.f), 20.f);
                    int k0 = kb + kc + r16;
                    if (k0      > qrow) s0 = -1e9f;
                    if (k0 + 16 > qrow) s1 = -1e9f;
                    float tmax = fmaxf(s0, s1);
#pragma unroll
                    for (int d = 8; d >= 1; d >>= 1)
                        tmax = fmaxf(tmax, __shfl_xor(tmax, d, 32));
                    float mnew  = fmaxf(mrow[p], tmax);
                    float scale = __expf(mrow[p] - mnew);
                    float p0 = __expf(s0 - mnew);
                    float p1 = __expf(s1 - mnew);
                    float tsum = p0 + p1;
#pragma unroll
                    for (int d = 8; d >= 1; d >>= 1)
                        tsum += __shfl_xor(tsum, d, 32);
                    lrow[p] = lrow[p] * scale + tsum;
                    mrow[p] = mnew;
#pragma unroll
                    for (int t = 0; t < 4; ++t) o[t][p] *= scale;
                    pp[(8 * lh + p) * 32 + r16     ] = f32_to_bf16(p0);
                    pp[(8 * lh + p) * 32 + 16 + r16] = f32_to_bf16(p1);
                }

                // ---- P fragment (A layout) from wave-private LDS ----
                FragB16 pf;
                {
                    int ko1 = lh ? 8 : 0, ko2 = lh ? 24 : 16;
                    pf.q[0] = *(const uint4*)(&pp[r16 * 32 + ko1]);
                    pf.q[1] = *(const uint4*)(&pp[r16 * 32 + ko2]);
                }
                // ---- O += P V : preload 4 V-fragments, then 4 WMMAs ----
                FragB16 vf[4];
#pragma unroll
                for (int t = 0; t < 4; ++t) {
                    int hcol = t * 16 + r16;                  // B: lane covers N(=hd), elements cover K(=key)
#pragma unroll
                    for (int w = 0; w < 8; ++w) {
                        int key = kc + (w < 4 ? 0 : 16) + 8 * lh + 2 * (w & 3);
                        vf[t].u[w] = (unsigned)sV[key * LKV + hcol] | ((unsigned)sV[(key + 1) * LKV + hcol] << 16);
                    }
                }
#pragma unroll
                for (int t = 0; t < 4; ++t)
                    o[t] = __builtin_amdgcn_wmma_f32_16x16x32_bf16(
                        false, pf.v, false, vf[t].v, (short)0, o[t], false, false);
            }
        }
        __syncthreads();
    }

    // ---- normalize + store ctx [B,S,H*64] ----
#pragma unroll
    for (int p = 0; p < 8; ++p) {
        int srow = q0 + wave * 16 + 8 * lh + p;
        float inv = 1.0f / lrow[p];
        unsigned short* orow = Octx + ((size_t)(b * S + srow)) * (AH * AHD) + h * AHD;
#pragma unroll
        for (int t = 0; t < 4; ++t)
            orow[t * 16 + r16] = f32_to_bf16(o[t][p] * inv);
    }
}

// ---------------------------------------------------------------------------
extern "C" void kernel_launch(void* const* d_in, const int* in_sizes, int n_in,
                              void* d_out, int out_size, void* d_ws, size_t ws_size,
                              hipStream_t stream) {
    (void)in_sizes; (void)n_in; (void)out_size; (void)ws_size;
    const float* x    = (const float*)d_in[0];
    const float* wq   = (const float*)d_in[1];
    const float* wk   = (const float*)d_in[2];
    const float* wv   = (const float*)d_in[3];
    const float* wo   = (const float*)d_in[4];
    const float* cosb = (const float*)d_in[5];
    const float* sinb = (const float*)d_in[6];

    const int B = 2, S = 2048, D = 2048, H = 32, KVH = 8;
    const int M = B * S;

    char* ws = (char*)d_ws;
    unsigned short* xb  = (unsigned short*)ws; ws += (size_t)M * 2048 * 2;
    unsigned short* wqb = (unsigned short*)ws; ws += (size_t)2048 * 2048 * 2;
    unsigned short* wkb = (unsigned short*)ws; ws += (size_t)2048 * 512 * 2;
    unsigned short* wvb = (unsigned short*)ws; ws += (size_t)2048 * 512 * 2;
    unsigned short* wob = (unsigned short*)ws; ws += (size_t)2048 * 2048 * 2;
    float*          xq  = (float*)ws;          ws += (size_t)M * 2048 * 4;
    float*          xk  = (float*)ws;          ws += (size_t)M * 512 * 4;
    float*          xv  = (float*)ws;          ws += (size_t)M * 512 * 4;
    unsigned short* qb  = (unsigned short*)ws; ws += (size_t)M * 2048 * 2;
    unsigned short* kbb = (unsigned short*)ws; ws += (size_t)M * 512 * 2;
    unsigned short* vbb = (unsigned short*)ws; ws += (size_t)M * 512 * 2;
    unsigned short* ctx = (unsigned short*)ws; ws += (size_t)M * 2048 * 2;

    dim3 blk(256);
    // one-shot bf16 conversion of activations + weights
    cvt_bf16<<<(M * 2048 / 4 + 255) / 256, blk, 0, stream>>>(x,  xb,  M * 2048 / 4);
    cvt_bf16<<<(2048 * 2048 / 4 + 255) / 256, blk, 0, stream>>>(wq, wqb, 2048 * 2048 / 4);
    cvt_bf16<<<(2048 * 512 / 4 + 255) / 256, blk, 0, stream>>>(wk, wkb, 2048 * 512 / 4);
    cvt_bf16<<<(2048 * 512 / 4 + 255) / 256, blk, 0, stream>>>(wv, wvb, 2048 * 512 / 4);
    cvt_bf16<<<(2048 * 2048 / 4 + 255) / 256, blk, 0, stream>>>(wo, wob, 2048 * 2048 / 4);

    // QKV projections (bf16 x bf16 -> f32)
    gemm_bf16<<<dim3(2048 / BN, M / BM), blk, 0, stream>>>(xb, wqb, xq, M, 2048, D);
    gemm_bf16<<<dim3(512  / BN, M / BM), blk, 0, stream>>>(xb, wkb, xk, M, 512,  D);
    gemm_bf16<<<dim3(512  / BN, M / BM), blk, 0, stream>>>(xb, wvb, xv, M, 512,  D);

    // RoPE + transpose to head-major bf16
    int nq = B * S * H * 32;
    int nk = B * S * KVH * 32;
    rope_to_bf16<<<(nq + 255) / 256, blk, 0, stream>>>(xq, cosb, sinb, qb,  H,   S, 1, nq);
    rope_to_bf16<<<(nk + 255) / 256, blk, 0, stream>>>(xk, cosb, sinb, kbb, KVH, S, 1, nk);
    rope_to_bf16<<<(nk + 255) / 256, blk, 0, stream>>>(xv, cosb, sinb, vbb, KVH, S, 0, nk);

    // Flash attention
    flash_attn<<<dim3(S / QB, H, B), blk, 0, stream>>>(qb, kbb, vbb, ctx, S);

    // Output projection
    gemm_bf16<<<dim3(2048 / BN, M / BM), blk, 0, stream>>>(ctx, wob, (float*)d_out, M, 2048, 2048);
}
